// SpatialEmbLoss_11501922418909
// MI455X (gfx1250) — matrix operand order repaired and checked by reading, not compiled, not run on hardware
//
#include <hip/hip_runtime.h>
#include <hip/hip_bf16.h>
#include <math.h>

typedef __attribute__((ext_vector_type(16))) _Float16 v16h;
typedef __attribute__((ext_vector_type(8)))  float    v8f;

#define IMG_H 512
#define IMG_W 512
#define HW    (IMG_H * IMG_W)
#define NB    8
#define NI    16
#define BINS  1024

// workspace layout (float offsets)
#define OFF_STATS 0      // [8][16][5]  sumx,sumy,sumsig,count,sumsig2
#define OFF_BG    640    // [8] background seed^2 sum
#define OFF_DER   648    // [8][16][6]  cx,cy,sexp,count,present,var
#define OFF_PC    1416   // [8] present count
#define OFF_VAR   1424   // [8] var loss
#define OFF_IL    1432   // [8] instance (lovasz) loss
#define OFF_SI    1440   // [8] instance seed loss
#define WS_FLOATS 1448
#define OFF_PRE   4096   // [8][3][HW] precomputed ex, ey, sigmoid(seed)

static __device__ __forceinline__ float fast_rcp(float x) {
    return __builtin_amdgcn_rcpf(x);            // v_rcp_f32, no div expansion
}
static __device__ __forceinline__ float fast_sigmoid(float x) {
    return fast_rcp(1.0f + __expf(-x));
}
static __device__ __forceinline__ float fast_tanh(float x) {
    return 2.0f * fast_rcp(1.0f + __expf(-2.0f * x)) - 1.0f;
}

// ---------------------------------------------------------------------------
// Kernel 1: masked per-instance reductions as WMMA GEMM.
//   D[m][n] += sum_k  mask(inst==m+1 && label==1)[pixel k] * feature_n[pixel k]
//   features: n=0:x  n=1:y  n=2:sigma  n=3:1  n=4:sigma^2  (rows 5..15 zero)
// A (16x32 f16 masks) built register-only from 16 wave ballots; B (32x16 f16
// features) staged through a strictly per-wave LDS region, so ordering needs
// only the CDNA5 same-wave DS in-order guarantee + a compiler wave_barrier —
// no workgroup barriers in the hot loop.
// Grid: 8 images * 64 WGs, 256 threads (8 waves), each wave owns 512 pixels.
// ---------------------------------------------------------------------------
__global__ __launch_bounds__(256) void k_stats(const float* __restrict__ pred,
                                               const int* __restrict__ inst,
                                               const int* __restrict__ lab,
                                               float* __restrict__ ws)
{
    __shared__ _Float16 sfeat[8][16][32];  // per-wave staging, 16 feature rows
    __shared__ float sbg;

    const int b    = blockIdx.x >> 6;      // 64 WGs per image
    const int wg   = blockIdx.x & 63;
    const int wave = threadIdx.x >> 5;
    const int lane = threadIdx.x & 31;
    const int waveBase = wg * 4096 + wave * 512;
    const float INV = 1.0f / 1023.0f;      // linspace(0,1,1024) step

    const float* sigp  = pred + ((size_t)b * 4 + 2) * HW;
    const float* seedp = pred + ((size_t)b * 4 + 3) * HW;
    const int*   ip    = inst + (size_t)b * HW;
    const int*   lp    = lab  + (size_t)b * HW;

    if (threadIdx.x == 0) sbg = 0.0f;
    __syncthreads();

    _Float16* fh = &sfeat[wave][0][0];
    unsigned* fw = (unsigned*)fh;
    const unsigned* fview = (const unsigned*)fh;

    // zero my wave's staging region once (256 dwords / 32 lanes = 8 each);
    // rows 5..15 stay zero forever -> B columns 5..15 contribute nothing.
#pragma unroll
    for (int j = 0; j < 8; ++j) fw[j * 32 + lane] = 0u;
    __builtin_amdgcn_wave_barrier();

    v8f c = {};
    float bga = 0.0f;

    const unsigned mrow = lane & 15;                   // A row (instance)
    const unsigned kbA  = (lane & 16) ? 16u : 0u;      // A K-group base
    const unsigned nB   = lane & 15;                   // B column (feature)
    const unsigned kwB  = (lane & 16) ? 8u : 0u;       // B K dword base

    for (int s = 0; s < 16; ++s) {
        const int p = waveBase + s * 32 + lane;        // pixel K = lane
        const float sg = sigp[p];
        const float sl = seedp[p];
        const int   iv = ip[p];
        const int   lv = lp[p];
        const float x  = (float)(p & 511) * INV;
        const float y  = (float)(p >> 9) * INV;

        // branchless background seed^2 accumulation
        const float sd = fast_sigmoid(sl);
        bga += (lv == 0) ? sd * sd : 0.0f;

        // 16 ballots: bit k of bm[m] = mask(instance m) at pixel k
        unsigned bm[16];
        const bool on = (lv == 1);
#pragma unroll
        for (int m = 0; m < 16; ++m)
            bm[m] = __builtin_amdgcn_ballot_w32(on && (iv == m + 1));
        unsigned mymask = 0u;
#pragma unroll
        for (int m = 0; m < 16; ++m)
            mymask = (mrow == (unsigned)m) ? bm[m] : mymask;

        // A fragment: 16-bit A 16x32 layout (VGPR j packs K pair)
        union { unsigned u[8]; v16h h; } A;
#pragma unroll
        for (int j = 0; j < 8; ++j) {
            const unsigned k0 = kbA + 2u * j;
            const unsigned h0 = ((mymask >> k0)        & 1u) ? 0x3C00u : 0u;
            const unsigned h1 = ((mymask >> (k0 + 1u)) & 1u) ? 0x3C00u : 0u;
            A.u[j] = h0 | (h1 << 16);
        }

        // stage features for my pixel (column K = lane); same-wave DS ops
        // complete in order, wave_barrier only pins the compile-time order.
        fh[0 * 32 + lane] = (_Float16)x;
        fh[1 * 32 + lane] = (_Float16)y;
        fh[2 * 32 + lane] = (_Float16)sg;
        fh[3 * 32 + lane] = (_Float16)1.0f;
        fh[4 * 32 + lane] = (_Float16)(sg * sg);
        __builtin_amdgcn_wave_barrier();

        // gather B fragment (uniform, no divergence; rows >=5 read zeros)
        union { unsigned u[8]; v16h h; } Bm;
#pragma unroll
        for (int j = 0; j < 8; ++j) Bm.u[j] = fview[nB * 16 + kwB + j];
        __builtin_amdgcn_wave_barrier();

        // D = A(16x32 masks) x B(32x16 features) + C  (EXEC all ones here)
        c = __builtin_amdgcn_wmma_f32_16x16x32_f16(
                false, A.h, false, Bm.h, (short)0, c, false, false);
    }

    // C layout: lane holds N = lane%16, rows M = r + (lane>=16 ? 8 : 0)
    if (nB < 5) {
        const int mofs = (lane & 16) ? 8 : 0;
#pragma unroll
        for (int r = 0; r < 8; ++r)
            atomicAdd(&ws[OFF_STATS + ((b * 16) + (r + mofs)) * 5 + (int)nB], c[r]);
    }

    atomicAdd(&sbg, bga);
    __syncthreads();
    if (threadIdx.x == 0) atomicAdd(&ws[OFF_BG + b], sbg);
}

// ---------------------------------------------------------------------------
// Kernel 2: per-(image,instance) finalize: center, s_mean, var, exp(10*s_mean)
// ---------------------------------------------------------------------------
__global__ void k_final_stats(float* __restrict__ ws)
{
    const int t = threadIdx.x;           // 128 threads = 8 images * 16 inst
    const int b = t >> 4, k = t & 15;
    const float* st = &ws[OFF_STATS + (b * 16 + k) * 5];
    const float sx = st[0], sy = st[1], ss = st[2], cnt = st[3], ss2 = st[4];
    const float rs = fast_rcp(fmaxf(cnt, 1.0f));
    const float cx = sx * rs, cy = sy * rs, sm = ss * rs;
    // sum m*(sigma - s_mean)^2 = (E[s^2] - 2*sm*E[s] + sm^2*n) / n
    const float var = (ss2 - 2.0f * sm * ss + sm * sm * cnt) * rs;
    const float sexp = __expf(10.0f * sm);
    const int present = (cnt > 0.0f);

    float* dv = &ws[OFF_DER + (b * 16 + k) * 6];
    dv[0] = cx; dv[1] = cy; dv[2] = sexp; dv[3] = cnt;
    dv[4] = present ? 1.0f : 0.0f; dv[5] = var;
    if (present) {
        atomicAdd(&ws[OFF_PC  + b], 1.0f);
        atomicAdd(&ws[OFF_VAR + b], var);
    }
}

// ---------------------------------------------------------------------------
// Kernel 2b: precompute instance-independent per-pixel planes once:
//   ex = tanh(pred0)+x,  ey = tanh(pred1)+y,  ssd = sigmoid(pred3)
// Removes the 16x redundant transcendental work from the dist kernel.
// ---------------------------------------------------------------------------
__global__ __launch_bounds__(256) void k_pre(const float* __restrict__ pred,
                                             float* __restrict__ pre)
{
    const int idx = blockIdx.x * 256 + threadIdx.x;   // NB*HW threads
    const int b = idx >> 18;                          // HW = 2^18
    const int p = idx & (HW - 1);
    const float INV = 1.0f / 1023.0f;
    const float* base = pred + (size_t)b * 4 * HW;
    __builtin_prefetch(&base[p + 8192], 0, 1);
    const float ex = fast_tanh(base[p])          + (float)(p & 511) * INV;
    const float ey = fast_tanh(base[HW + p])     + (float)(p >> 9) * INV;
    const float sd = fast_sigmoid(base[3 * HW + p]);
    float* pb = pre + (size_t)b * 3 * HW;
    pb[p]          = ex;
    pb[HW + p]     = ey;
    pb[2 * HW + p] = sd;
}

// ---------------------------------------------------------------------------
// Kernel 3: one WG per (image, instance). Computes dist over all pixels,
// builds a 1024-bin LDS error histogram (pos/neg split), accumulates the
// instance seed loss, then does the closed-form Lovasz scan over bins:
//   positive run: union invariant -> grad = 1/U each
//   negative run: grads telescope  -> I*(1/U - 1/(U+n))
// ---------------------------------------------------------------------------
__global__ __launch_bounds__(256) void k_inst(const float* __restrict__ pred,
                                              const int* __restrict__ inst,
                                              const int* __restrict__ lab,
                                              float* __restrict__ ws,
                                              const float* __restrict__ pre,
                                              const int usePre)
{
    __shared__ unsigned h_np[BINS];
    __shared__ float    h_sp[BINS];
    __shared__ unsigned h_nn[BINS];
    __shared__ float    h_sn[BINS];
    __shared__ float    ssum;

    const int b = blockIdx.x >> 4;
    const int k = blockIdx.x & 15;
    const float* dv = &ws[OFF_DER + (b * 16 + k) * 6];
    if (dv[4] == 0.0f) return;           // instance absent: contributes zero
    const float cx = dv[0], cy = dv[1], sexp = dv[2], G = dv[3];

    for (int i = threadIdx.x; i < BINS; i += 256) {
        h_np[i] = 0u; h_sp[i] = 0.0f; h_nn[i] = 0u; h_sn[i] = 0.0f;
    }
    if (threadIdx.x == 0) ssum = 0.0f;
    __syncthreads();

    const float* e0p = pred + ((size_t)b * 4 + 0) * HW;
    const float* e1p = pred + ((size_t)b * 4 + 1) * HW;
    const float* sdp = pred + ((size_t)b * 4 + 3) * HW;
    const float* pxp = pre + (size_t)b * 3 * HW;
    const float* pyp = pxp + HW;
    const float* psp = pxp + 2 * HW;
    const int*   ip  = inst + (size_t)b * HW;
    const int*   lp  = lab  + (size_t)b * HW;
    const float INV = 1.0f / 1023.0f;

    float sacc = 0.0f;
    for (int p = threadIdx.x; p < HW; p += 256) {
        float ex, ey;
        if (usePre) {
            __builtin_prefetch(&pxp[p + 4096], 0, 1);
            __builtin_prefetch(&pyp[p + 4096], 0, 1);
            ex = pxp[p];
            ey = pyp[p];
        } else {
            ex = fast_tanh(e0p[p]) + (float)(p & 511) * INV;
            ey = fast_tanh(e1p[p]) + (float)(p >> 9) * INV;
        }
        const float dx = ex - cx, dy = ey - cy;
        const float d  = __expf(-sexp * (dx * dx + dy * dy));
        const bool pos = (ip[p] == k + 1) && (lp[p] == 1);
        // logits = 2d-1, signs = +/-1 -> errors in [0,2] (relu is identity)
        const float err = pos ? (2.0f - 2.0f * d) : (2.0f * d);
        int bin = (int)(err * (float)(BINS / 2));
        bin = bin < 0 ? 0 : (bin > BINS - 1 ? BINS - 1 : bin);
        if (pos) {
            atomicAdd(&h_np[bin], 1u);
            atomicAdd(&h_sp[bin], err);
            const float sd = usePre ? psp[p] : fast_sigmoid(sdp[p]);
            const float df = sd - d;
            sacc += df * df;
        } else {
            atomicAdd(&h_nn[bin], 1u);
            atomicAdd(&h_sn[bin], err);
        }
    }
    atomicAdd(&ssum, sacc);
    __syncthreads();

    if (threadIdx.x == 0) {
        atomicAdd(&ws[OFF_SI + b], ssum);
        // Lovasz scan, descending error
        float cgt = 0.0f, j = 0.0f, loss = 0.0f;
        for (int i = BINS - 1; i >= 0; --i) {
            const float np = (float)h_np[i];
            if (np > 0.0f) {
                const float U = G + j - cgt;           // invariant over pos run
                loss += h_sp[i] * fast_rcp(U);
                cgt += np; j += np;
            }
            const float nn = (float)h_nn[i];
            if (nn > 0.0f) {
                const float I  = G - cgt;
                const float U0 = G + j - cgt;
                loss += (h_sn[i] * fast_rcp(nn)) * I *
                        (fast_rcp(U0) - fast_rcp(U0 + nn));
                j += nn;
            }
        }
        atomicAdd(&ws[OFF_IL + b], loss);
    }
}

// ---------------------------------------------------------------------------
// Kernel 4: combine per-image terms, mean over batch.
// ---------------------------------------------------------------------------
__global__ void k_reduce(const float* __restrict__ ws, float* __restrict__ out)
{
    if (threadIdx.x == 0) {
        float tot = 0.0f;
        for (int b = 0; b < NB; ++b) {
            const float obj  = fmaxf(ws[OFF_PC + b], 1.0f);
            const float seed = ws[OFF_BG + b] + ws[OFF_SI + b];
            tot += 1.0f  * ws[OFF_IL  + b] / obj
                 + 10.0f * ws[OFF_VAR + b] / obj
                 + 1.0f  * seed / (float)HW;
        }
        out[0] = tot / (float)NB;
    }
}

extern "C" void kernel_launch(void* const* d_in, const int* in_sizes, int n_in,
                              void* d_out, int out_size, void* d_ws, size_t ws_size,
                              hipStream_t stream) {
    const float* pred = (const float*)d_in[0];
    const int*   inst = (const int*)d_in[1];
    const int*   lab  = (const int*)d_in[2];
    float* ws = (float*)d_ws;
    float* pre = ws + OFF_PRE;

    const size_t needed = ((size_t)OFF_PRE + (size_t)NB * 3 * HW) * sizeof(float);
    const int usePre = (ws_size >= needed) ? 1 : 0;

    hipMemsetAsync(d_ws, 0, WS_FLOATS * sizeof(float), stream);
    k_stats<<<NB * 64, 256, 0, stream>>>(pred, inst, lab, ws);
    k_final_stats<<<1, 128, 0, stream>>>(ws);
    if (usePre)
        k_pre<<<(NB * HW) / 256, 256, 0, stream>>>(pred, pre);
    k_inst<<<NB * NI, 256, 0, stream>>>(pred, inst, lab, ws, pre, usePre);
    k_reduce<<<1, 32, 0, stream>>>(ws, (float*)d_out);
}
